// SentRNN_7945689497864
// MI455X (gfx1250) — matrix operand order repaired and checked by reading, not compiled, not run on hardware
//
#include <hip/hip_runtime.h>

typedef __attribute__((ext_vector_type(16))) _Float16 v16h;
typedef __attribute__((ext_vector_type(8)))  _Float16 v8h;
typedef __attribute__((ext_vector_type(8)))  float    v8f;
typedef __attribute__((ext_vector_type(4)))  float    v4f;

#define SEQ   2048
#define BATCH 64
#define ISZ   256
#define HID   128

static __device__ __forceinline__ v8f wmma32(v16h a, v16h b, v8f c) {
  // D = A(16x32 f16) * B(32x16 f16) + C(16x16 f32)
  return __builtin_amdgcn_wmma_f32_16x16x32_f16(false, a, false, b, (short)0, c,
                                                false, false);
}

static __device__ __forceinline__ float fast_sigmoid(float x) {
  return __builtin_amdgcn_rcpf(1.0f + __expf(-x));
}
static __device__ __forceinline__ float fast_tanh(float x) {
  return 1.0f - 2.0f * __builtin_amdgcn_rcpf(1.0f + __expf(2.0f * x));
}

// A fragment (16x32 f16 tile): lane row = tm + (lane&15);
// K elements: [k+0..7] -> v[0..7], [k+16..23] -> v[8..15]; caller adds koffA=(lane>>4)*8.
static __device__ __forceinline__ v16h fragA_f16(const _Float16* rp, int k) {
  v8h lo = *(const v8h*)(rp + k);
  v8h hi = *(const v8h*)(rp + k + 16);
  v16h f;
#pragma unroll
  for (int e = 0; e < 8; ++e) { f[e] = lo[e]; f[e + 8] = hi[e]; }
  return f;
}
static __device__ __forceinline__ v16h fragA_f32(const float* rp, int k) {
  v4f a0 = *(const v4f*)(rp + k);
  v4f a1 = *(const v4f*)(rp + k + 4);
  v4f b0 = *(const v4f*)(rp + k + 16);
  v4f b1 = *(const v4f*)(rp + k + 20);
  v16h f;
#pragma unroll
  for (int e = 0; e < 4; ++e) {
    f[e]      = (_Float16)a0[e];
    f[e + 4]  = (_Float16)a1[e];
    f[e + 8]  = (_Float16)b0[e];
    f[e + 12] = (_Float16)b1[e];
  }
  return f;
}
// B fragment (32x16 f16 tile) from f32 weights stored (N x K) row-major:
// lane col n = tn + (lane&15); K elements [k+0..7],[k+16..23]; caller adds koffB=(lane>>4)*8.
static __device__ __forceinline__ v16h fragB_f32(const float* rp, int k) {
  v4f a0 = *(const v4f*)(rp + k);
  v4f a1 = *(const v4f*)(rp + k + 4);
  v4f a2 = *(const v4f*)(rp + k + 16);
  v4f a3 = *(const v4f*)(rp + k + 20);
  v16h f;
#pragma unroll
  for (int e = 0; e < 4; ++e) {
    f[e]      = (_Float16)a0[e];
    f[e + 4]  = (_Float16)a1[e];
    f[e + 8]  = (_Float16)a2[e];
    f[e + 12] = (_Float16)a3[e];
  }
  return f;
}

// Generic Y = X @ W^T (+bias): X is (M x K), W is (N x K) row-major.
// One wave per 16x16 output tile. N = ntx*16. Grid sized exactly (8 waves/block).
template <bool A_HALF, bool HAS_BIAS, bool OUT32, bool OUT16>
__global__ __launch_bounds__(256) void gemm16(
    const void* __restrict__ Ap, const float* __restrict__ W,
    const float* __restrict__ bias, float* __restrict__ Y32,
    _Float16* __restrict__ Y16, int K, int ntx) {
  const int lane  = threadIdx.x & 31;
  const int wid   = blockIdx.x * 8 + (threadIdx.x >> 5);
  const int tm    = (wid / ntx) << 4;
  const int tn    = (wid % ntx) << 4;
  const int r15   = lane & 15;
  const int hi8   = (lane >> 4) << 3;
  const int koffA = hi8;
  const int koffB = (lane >> 4) << 3;
  const int N     = ntx << 4;

  v8f acc = {};
  const float* wrow = W + (size_t)(tn + r15) * K;
  if constexpr (A_HALF) {
    const _Float16* arow = (const _Float16*)Ap + (size_t)(tm + r15) * K;
    for (int k = 0; k < K; k += 32)
      acc = wmma32(fragA_f16(arow, k + koffA), fragB_f32(wrow, k + koffB), acc);
  } else {
    const float* arow = (const float*)Ap + (size_t)(tm + r15) * K;
    for (int k = 0; k < K; k += 32)
      acc = wmma32(fragA_f32(arow, k + koffA), fragB_f32(wrow, k + koffB), acc);
  }
  float bv = 0.0f;
  if constexpr (HAS_BIAS) bv = bias[tn + r15];
#pragma unroll
  for (int r = 0; r < 8; ++r) {
    size_t idx = (size_t)(tm + r + hi8) * N + (tn + r15);
    float v = acc[r] + bv;
    if constexpr (OUT32) Y32[idx] = v;
    if constexpr (OUT16) Y16[idx] = (_Float16)v;
  }
}

// Per-step x-gate pre-activations for this lane (register double buffer).
struct XBuf {
  float r[8], z[8], n[8];
};

// Persistent bidirectional GRU scan. Batch rows are independent
// (h @ w_hh^T mixes only the hidden dim): 8 blocks = 2 dirs x 4 batch
// quarters, 256 threads = 8 waves. Wave w owns hidden tile jt=w*16 and the
// r/z/n gate tiles for the same (b,j) => no cross-wave gate exchange.
// Latency hiding in the serial loop:
//   - w_hh B-fragments (12 x v16h) preloaded to registers once
//   - xg slice for step t+1 register-prefetched, t+4 cache-prefetched
//   - h16 double-buffered in LDS => single barrier per step
__global__ __launch_bounds__(256, 1) void gru_scan(
    const float* __restrict__ xg_f, const float* __restrict__ xg_b,
    const float* __restrict__ whh_f, const float* __restrict__ whh_b,
    const float* __restrict__ bhh_f, const float* __restrict__ bhh_b,
    const float* __restrict__ hid0, float* __restrict__ out32,
    _Float16* __restrict__ out16, float* __restrict__ hT) {
  const int dir = blockIdx.x >> 2;   // 0..1
  const int mq  = blockIdx.x & 3;    // batch quarter
  const int mt  = mq << 4;           // global batch row base (16 rows)
  const float* xg  = dir ? xg_b : xg_f;
  const float* whh = dir ? whh_b : whh_f;
  const float* bhh = dir ? bhh_b : bhh_f;
  const float* h0  = hid0 + (size_t)dir * BATCH * HID + (size_t)mt * HID;
  float* hTd       = hT + (size_t)dir * BATCH * HID + (size_t)mt * HID;

  __shared__ float    h32[16 * HID];    // 8 KB master state (this quarter)
  __shared__ _Float16 h16a[16 * HID];   // 4 KB WMMA feed, ping
  __shared__ _Float16 h16b[16 * HID];   // 4 KB WMMA feed, pong

  const int tid = threadIdx.x;
  for (int i = tid; i < 16 * HID; i += 256) {
    float v = h0[i];
    h32[i]  = v;
    h16a[i] = (_Float16)v;
  }
  __syncthreads();

  const int lane  = tid & 31;
  const int w     = tid >> 5;        // 0..7 -> hidden tile
  const int jt    = w << 4;
  const int r15   = lane & 15;
  const int hi8   = (lane >> 4) << 3;
  const int koffA = hi8;
  const int koffB = (lane >> 4) << 3;
  const int jcol  = jt + r15;

  const float bhr = bhh[jcol];
  const float bhz = bhh[jcol + HID];
  const float bhn = bhh[jcol + 2 * HID];

  // Preload loop-invariant w_hh fragments into registers (f32 -> f16 once).
  const float* wrp = whh + (size_t)jcol * HID;
  const float* wzp = whh + (size_t)(jcol + HID) * HID;
  const float* wnp = whh + (size_t)(jcol + 2 * HID) * HID;
  v16h Br[4], Bz[4], Bn[4];
#pragma unroll
  for (int kk = 0; kk < 4; ++kk) {
    Br[kk] = fragB_f32(wrp, kk * 32 + koffB);
    Bz[kk] = fragB_f32(wzp, kk * 32 + koffB);
    Bn[kk] = fragB_f32(wnp, kk * 32 + koffB);
  }

  // Additively-advanced global pointers (no per-step 64-bit mul).
  const int s0 = dir ? (SEQ - 1) : 0;
  const ptrdiff_t xstep = dir ? -(ptrdiff_t)(BATCH * 3 * HID) : (BATCH * 3 * HID);
  const ptrdiff_t ostep = dir ? -(ptrdiff_t)(BATCH * 2 * HID) : (BATCH * 2 * HID);
  const float* xp0 =
      xg + (size_t)s0 * BATCH * 3 * HID + (size_t)(mt + hi8) * 3 * HID + jcol;
  // cache-prefetch pointer: this block's contiguous 24 KB slice, 4 steps ahead
  const float* pfbase = xg + ((size_t)s0 * BATCH + mt) * (3 * HID) + 4 * xstep;
  size_t obase = ((size_t)s0 * BATCH + (mt + hi8)) * (2 * HID) +
                 (size_t)dir * HID + jcol;
  float*    op32 = out32 + obase;
  _Float16* op16 = out16 + obase;

  auto load_xbuf = [&](XBuf& x, const float* base) {
#pragma unroll
    for (int r = 0; r < 8; ++r) {
      const float* xp = base + r * (3 * HID);
      x.r[r] = xp[0];
      x.z[r] = xp[HID];
      x.n[r] = xp[2 * HID];
    }
  };

  // One GRU step: WMMA from h16r, gates from xc, new state -> h32/h16w,
  // while register-prefetching next step's x into xnx.
  auto gru_step = [&](const _Float16* h16r, _Float16* h16w, const XBuf& xc,
                      XBuf& xnx) {
    // prefetch step t+1 (registers) and t+4 (cache); independent of h
    load_xbuf(xnx, xp0 + xstep);
    __builtin_prefetch((const char*)pfbase + tid * 96, 0, 1);

    const _Float16* arow = h16r + (size_t)r15 * HID;
    v8f ar = {}, az = {}, an = {};
#pragma unroll
    for (int kk = 0; kk < 4; ++kk) {
      v16h a = fragA_f16(arow, kk * 32 + koffA);
      ar = wmma32(a, Br[kk], ar);
      az = wmma32(a, Bz[kk], az);
      an = wmma32(a, Bn[kk], an);
    }
    // gate math writes the OTHER h16 buffer -> no barrier needed here
#pragma unroll
    for (int r = 0; r < 8; ++r) {
      const int bl = r + hi8;  // local batch row 0..15 (D layout)
      float rg   = fast_sigmoid(xc.r[r] + ar[r] + bhr);
      float zg   = fast_sigmoid(xc.z[r] + az[r] + bhz);
      float ng   = fast_tanh(xc.n[r] + rg * (an[r] + bhn));
      float hold = h32[bl * HID + jcol];  // owner-lane only: race-free
      float hnew = (1.0f - zg) * ng + zg * hold;
      h32[bl * HID + jcol] = hnew;
      h16w[bl * HID + jcol] = (_Float16)hnew;
      op32[r * (2 * HID)] = hnew;
      op16[r * (2 * HID)] = (_Float16)hnew;
    }
    xp0 += xstep;
    pfbase += xstep;
    op32 += ostep;
    op16 += ostep;
    __syncthreads();  // h16w visible before next step's WMMA reads
  };

  XBuf xa, xb;
  load_xbuf(xa, xp0);
  for (int t = 0; t < SEQ; t += 2) {
    gru_step(h16a, h16b, xa, xb);  // even step: read ping, write pong
    gru_step(h16b, h16a, xb, xa);  // odd  step: read pong, write ping
  }
  for (int i = tid; i < 16 * HID; i += 256) hTd[i] = h32[i];
}

// softmax over the batch axis of t (S,B,2H), in place. One block per s,
// one thread per channel j; reads are coalesced across j.
__global__ __launch_bounds__(256) void softmax_batch(float* __restrict__ t32) {
  const int s = blockIdx.x;
  const int j = threadIdx.x;
  float v[BATCH];
  float mx = -3.0e38f;
#pragma unroll
  for (int b = 0; b < BATCH; ++b) {
    v[b] = t32[((size_t)s * BATCH + b) * (2 * HID) + j];
    mx = fmaxf(mx, v[b]);
  }
  float se = 0.0f;
#pragma unroll
  for (int b = 0; b < BATCH; ++b) {
    v[b] = __expf(v[b] - mx);
    se += v[b];
  }
  const float inv = __builtin_amdgcn_rcpf(se);
#pragma unroll
  for (int b = 0; b < BATCH; ++b)
    t32[((size_t)s * BATCH + b) * (2 * HID) + j] = v[b] * inv;
}

// sent[b,j] = sum_s attn[s,b,j]*out[s,b,j]; split over 64 s-chunks of 32.
__global__ __launch_bounds__(256) void wsum_partial(
    const float* __restrict__ attn, const float* __restrict__ o32,
    float* __restrict__ partial) {
  const int bj = blockIdx.x * 256 + threadIdx.x;  // 0..16383
  const int c  = blockIdx.y;                      // 0..63
  const int b  = bj >> 8, j = bj & 255;
  float acc = 0.0f;
  for (int s = c * 32; s < c * 32 + 32; ++s) {
    size_t idx = ((size_t)s * BATCH + b) * (2 * HID) + j;
    acc += attn[idx] * o32[idx];
  }
  partial[(size_t)c * 16384 + bj] = acc;
}

__global__ __launch_bounds__(256) void wsum_final(
    const float* __restrict__ partial, float* __restrict__ sent) {
  const int bj = blockIdx.x * 256 + threadIdx.x;
  float acc = 0.0f;
#pragma unroll
  for (int c = 0; c < 64; ++c) acc += partial[(size_t)c * 16384 + bj];
  sent[bj] = acc;
}

extern "C" void kernel_launch(void* const* d_in, const int* in_sizes, int n_in,
                              void* d_out, int out_size, void* d_ws,
                              size_t ws_size, hipStream_t stream) {
  (void)in_sizes; (void)n_in; (void)out_size; (void)ws_size;
  const float* inp    = (const float*)d_in[0];
  const float* hid0   = (const float*)d_in[1];
  const float* wih_f  = (const float*)d_in[2];
  const float* whh_f  = (const float*)d_in[3];
  const float* bih_f  = (const float*)d_in[4];
  const float* bhh_f  = (const float*)d_in[5];
  const float* wih_b  = (const float*)d_in[6];
  const float* whh_b  = (const float*)d_in[7];
  const float* bih_b  = (const float*)d_in[8];
  const float* bhh_b  = (const float*)d_in[9];
  const float* attn_w = (const float*)d_in[10];
  const float* attn_b = (const float*)d_in[11];
  const float* comb_w = (const float*)d_in[12];
  float* out = (float*)d_out;

  // workspace layout (phase-disjoint regions aliased to cap peak at ~604 MB)
  char* ws = (char*)d_ws;
  const size_t SB = (size_t)SEQ * BATCH;                    // 131072 rows
  float*    xg_f  = (float*)(ws);                           // SB*384 f32
  float*    xg_b  = (float*)(ws + SB * 384 * 4);            // SB*384 f32
  float*    o32   = (float*)(ws + 2 * SB * 384 * 4);        // SB*256 f32
  _Float16* o16   = (_Float16*)(ws + 2 * SB * 384 * 4 + SB * 256 * 4);
  _Float16* ann16 = (_Float16*)xg_b;         // alias: xg_b dead after scan
  float*    t32   = xg_f;                    // alias: xg_f dead after scan
  float*    part  = (float*)(ws + SB * 384 * 4 + SB * 256 * 2);  // in xg_b region

  dim3 blk(256);
  // 1) xg = inp @ w_ih^T + b_ih, both dirs: M=SB, K=256, N=384 (ntx=24)
  const int blocksA = (int)((SB / 16) * 24 / 8);  // 24576
  gemm16<false, true, true, false><<<blocksA, blk, 0, stream>>>(
      inp, wih_f, bih_f, xg_f, nullptr, ISZ, 24);
  gemm16<false, true, true, false><<<blocksA, blk, 0, stream>>>(
      inp, wih_b, bih_b, xg_b, nullptr, ISZ, 24);

  // 2) persistent bidirectional scan (8 blocks: 2 dirs x 4 batch-quarters)
  gru_scan<<<8, 256, 0, stream>>>(xg_f, xg_b, whh_f, whh_b, bhh_f, bhh_b,
                                  hid0, o32, o16, out + 16384);

  // 3) ann = out_state @ attn_w^T + attn_b  (f16 output feeds next GEMM)
  const int blocksB = (int)((SB / 16) * 16 / 8);  // 16384
  gemm16<true, true, false, true><<<blocksB, blk, 0, stream>>>(
      o16, attn_w, attn_b, nullptr, ann16, 2 * HID, 16);
  // 4) t = ann @ comb_w^T
  gemm16<true, false, true, false><<<blocksB, blk, 0, stream>>>(
      ann16, comb_w, nullptr, t32, nullptr, 2 * HID, 16);

  // 5) softmax over batch axis (faithful to reference), in place
  softmax_batch<<<SEQ, 256, 0, stream>>>(t32);

  // 6) sent = sum_s attn * out_state  -> d_out[0..16383]
  dim3 g1(64, 64);
  wsum_partial<<<g1, blk, 0, stream>>>(t32, o32, part);
  wsum_final<<<64, blk, 0, stream>>>(part, out);
}
